// MoeKANLayer_4801773437388
// MI455X (gfx1250) — compile-verified, matrix operand order checked by reading, not run on hardware
//
#include <hip/hip_runtime.h>

// ---- problem constants (match reference setup_inputs) ----
#define NB   4096   // B*S tokens
#define HD   512    // hidden
#define FD   1024   // ffn dim
#define NE   8      // experts

typedef __attribute__((ext_vector_type(16))) __bf16 v16bf;
typedef __attribute__((ext_vector_type(8)))  float  v8f;

// uniform grid from make_grid(): g(j) = 0.4*(j-3) - 1, j = 0..11
__device__ __forceinline__ float gridv(int j) {
    return 0.4f * (float)(j - 3) - 1.0f;
}

// Cox-de Boor order-3 -> 8 bases (identical recursion to reference; uniform denoms = 0.4k)
__device__ __forceinline__ void bspline8(float x, float* b) {
    float t[11];
#pragma unroll
    for (int i = 0; i < 11; ++i)
        t[i] = (x >= gridv(i) && x < gridv(i + 1)) ? 1.0f : 0.0f;
#pragma unroll
    for (int k = 1; k <= 3; ++k) {
        const float inv = 1.0f / (0.4f * (float)k);
#pragma unroll
        for (int i = 0; i < 11 - k; ++i)
            t[i] = (x - gridv(i)) * inv * t[i] + (gridv(i + k + 1) - x) * inv * t[i + 1];
    }
#pragma unroll
    for (int i = 0; i < 8; ++i) b[i] = t[i];
}

__device__ __forceinline__ float silu_f(float x) {
    return x / (1.0f + __expf(-x));
}

// ---- WMMA fragment loader (bf16 16x16x32; lane idx=lane&15, K groups by lane>>4) ----
__device__ __forceinline__ v16bf load_frag(const __bf16* rowp, int koff) {
    union { v16bf v; uint4 q[2]; } u;
    u.q[0] = *(const uint4*)(rowp + koff);        // k = g*8 .. g*8+7
    u.q[1] = *(const uint4*)(rowp + koff + 16);   // k = 16+g*8 .. +7
    return u.v;
}

__device__ __forceinline__ v8f wmma_bf16(v16bf a, v16bf b, v8f c) {
    return __builtin_amdgcn_wmma_f32_16x16x32_bf16(false, a, false, b, (short)0, c,
                                                   false, false);
}

// ---- fp32 -> bf16 weight pre-conversion (8 elements / thread) ----
__global__ __launch_bounds__(256) void cvt_kernel(
    const float* __restrict__ src, __bf16* __restrict__ dst, int n8)
{
    const int i = blockIdx.x * blockDim.x + threadIdx.x;
    if (i >= n8) return;
    const float4* s = (const float4*)src + (size_t)i * 2;
    const float4 f0 = s[0], f1 = s[1];
    union { __bf16 e[8]; uint4 q; } pk;
    pk.e[0] = (__bf16)f0.x; pk.e[1] = (__bf16)f0.y;
    pk.e[2] = (__bf16)f0.z; pk.e[3] = (__bf16)f0.w;
    pk.e[4] = (__bf16)f1.x; pk.e[5] = (__bf16)f1.y;
    pk.e[6] = (__bf16)f1.z; pk.e[7] = (__bf16)f1.w;
    ((uint4*)dst)[i] = pk.q;
}

// ---- gating: logits, top-2, softmax, scatter to per-expert row lists ----
__global__ __launch_bounds__(256) void gate_kernel(
    const float* __restrict__ x, const float* __restrict__ gw,
    int* __restrict__ cnt, int* __restrict__ rows, float* __restrict__ coef)
{
    const int n = blockIdx.x * blockDim.x + threadIdx.x;
    if (n >= NB) return;
    const float4* xp = (const float4*)(x + (size_t)n * HD);
    float acc[NE];
#pragma unroll
    for (int e = 0; e < NE; ++e) acc[e] = 0.0f;
    for (int h4 = 0; h4 < HD / 4; ++h4) {
        const float4 xv = xp[h4];
#pragma unroll
        for (int e = 0; e < NE; ++e) {
            const float4 wv = ((const float4*)(gw + (size_t)e * HD))[h4];
            acc[e] += xv.x * wv.x + xv.y * wv.y + xv.z * wv.z + xv.w * wv.w;
        }
    }
    int e0 = 0; float l0 = acc[0];
#pragma unroll
    for (int e = 1; e < NE; ++e) if (acc[e] > l0) { l0 = acc[e]; e0 = e; }
    int e1 = -1; float l1 = -3.4e38f;
#pragma unroll
    for (int e = 0; e < NE; ++e) if (e != e0 && acc[e] > l1) { l1 = acc[e]; e1 = e; }
    const float w0 = 1.0f / (1.0f + __expf(l1 - l0));
    const float w1 = 1.0f - w0;
    int p0 = atomicAdd(&cnt[e0], 1);
    rows[(size_t)e0 * NB + p0] = n;  coef[(size_t)e0 * NB + p0] = w0;
    int p1 = atomicAdd(&cnt[e1], 1);
    rows[(size_t)e1 * NB + p1] = n;  coef[(size_t)e1 * NB + p1] = w1;
}

// ---- precompute silu(x) and 8 spline bases per (token,feature), bf16 ----
__global__ __launch_bounds__(256) void prep_x_kernel(
    const float* __restrict__ x, __bf16* __restrict__ silu_x, __bf16* __restrict__ bases_x)
{
    const int idx = blockIdx.x * blockDim.x + threadIdx.x;
    if (idx >= NB * HD) return;
    const float v = x[idx];
    silu_x[idx] = (__bf16)silu_f(v);
    float b[8]; bspline8(v, b);
    union { __bf16 e[8]; uint4 q; } pk;
#pragma unroll
    for (int c = 0; c < 8; ++c) pk.e[c] = (__bf16)b[c];
    *(uint4*)(bases_x + (size_t)idx * 8) = pk.q;
}

// ---- layers 1&2 fused: h = kan1(x) * kan2(x); epilogue emits silu(h), bspl(h) ----
// block = 8 waves, tile 64(M) x 64(N); wave = 32(M, two subtiles) x 16(N), 4 accumulators
__global__ __launch_bounds__(256) void kan12_kernel(
    const __bf16* __restrict__ silu_x, const __bf16* __restrict__ bases_x,
    const __bf16* __restrict__ bw1, const __bf16* __restrict__ sw1,
    const __bf16* __restrict__ bw2, const __bf16* __restrict__ sw2,
    const int* __restrict__ cnt_e, const int* __restrict__ rows_e,
    __bf16* __restrict__ silu_h, __bf16* __restrict__ bases_h)
{
    const int count = *cnt_e;
    const int wid = threadIdx.x >> 5, lane = threadIdx.x & 31;
    const int wm = wid & 1, wn = wid >> 1;                 // 2x4 waves
    const int mBase = blockIdx.x * 64 + wm * 32;
    const int nBase = blockIdx.y * 64 + wn * 16;
    if (mBase >= count) return;
    const int g = lane >> 4, li = lane & 15, g8 = g * 8;

    int r0 = mBase + li;      if (r0 >= count) r0 = count - 1;
    int r1 = mBase + 16 + li; if (r1 >= count) r1 = count - 1;
    const int t0 = rows_e[r0], t1 = rows_e[r1];
    const __bf16* a0b = silu_x  + (size_t)t0 * HD;
    const __bf16* a1b = silu_x  + (size_t)t1 * HD;
    const __bf16* a0s = bases_x + (size_t)t0 * (HD * 8);
    const __bf16* a1s = bases_x + (size_t)t1 * (HD * 8);

    const int f = nBase + li;
    const __bf16* w1b = bw1 + (size_t)f * HD;
    const __bf16* w2b = bw2 + (size_t)f * HD;
    const __bf16* w1s = sw1 + (size_t)f * (HD * 8);
    const __bf16* w2s = sw2 + (size_t)f * (HD * 8);

    v8f acc1a = {}, acc1b = {}, acc2a = {}, acc2b = {};
    for (int kb = 0; kb < HD; kb += 32) {                  // base path, K = H
        const int ko = kb + g8;
        v16bf a0 = load_frag(a0b, ko);
        v16bf a1 = load_frag(a1b, ko);
        v16bf b1 = load_frag(w1b, ko);
        acc1a = wmma_bf16(a0, b1, acc1a);
        acc1b = wmma_bf16(a1, b1, acc1b);
        v16bf b2 = load_frag(w2b, ko);
        acc2a = wmma_bf16(a0, b2, acc2a);
        acc2b = wmma_bf16(a1, b2, acc2b);
    }
    for (int kb = 0; kb < HD * 8; kb += 32) {              // spline path, K = 8H
        const int ko = kb + g8;
        v16bf a0 = load_frag(a0s, ko);
        v16bf a1 = load_frag(a1s, ko);
        v16bf b1 = load_frag(w1s, ko);
        acc1a = wmma_bf16(a0, b1, acc1a);
        acc1b = wmma_bf16(a1, b1, acc1b);
        v16bf b2 = load_frag(w2s, ko);
        acc2a = wmma_bf16(a0, b2, acc2a);
        acc2b = wmma_bf16(a1, b2, acc2b);
    }
    // C/D layout: lane holds column n = lane&15; VGPR v holds row m = v + 8*(lane>>4)
#pragma unroll
    for (int sub = 0; sub < 2; ++sub) {
        const v8f* p1 = sub ? &acc1b : &acc1a;
        const v8f* p2 = sub ? &acc2b : &acc2a;
#pragma unroll
        for (int v = 0; v < 8; ++v) {
            const int rr = mBase + sub * 16 + g8 + v;
            if (rr >= count) continue;
            const float h = (*p1)[v] * (*p2)[v];
            silu_h[(size_t)rr * FD + f] = (__bf16)silu_f(h);
            float bb[8]; bspline8(h, bb);
            union { __bf16 e[8]; uint4 q; } pk;
#pragma unroll
            for (int c = 0; c < 8; ++c) pk.e[c] = (__bf16)bb[c];
            *(uint4*)(bases_h + ((size_t)rr * FD + f) * 8) = pk.q;
        }
    }
}

// ---- layer 3: y = kan3(h); epilogue atomically accumulates coef * y into out ----
__global__ __launch_bounds__(256) void kan3_kernel(
    const __bf16* __restrict__ silu_h, const __bf16* __restrict__ bases_h,
    const __bf16* __restrict__ bw3, const __bf16* __restrict__ sw3,
    const int* __restrict__ cnt_e, const int* __restrict__ rows_e,
    const float* __restrict__ coef_e, float* __restrict__ out)
{
    const int count = *cnt_e;
    const int wid = threadIdx.x >> 5, lane = threadIdx.x & 31;
    const int wm = wid & 1, wn = wid >> 1;                 // 2x4 waves, 64x64 block tile
    const int mBase = blockIdx.x * 64 + wm * 32;
    const int nBase = blockIdx.y * 64 + wn * 16;
    if (mBase >= count) return;
    const int g = lane >> 4, li = lane & 15, g8 = g * 8;

    int r0 = mBase + li;      if (r0 >= count) r0 = count - 1;
    int r1 = mBase + 16 + li; if (r1 >= count) r1 = count - 1;
    const __bf16* a0b = silu_h  + (size_t)r0 * FD;
    const __bf16* a1b = silu_h  + (size_t)r1 * FD;
    const __bf16* a0s = bases_h + (size_t)r0 * (FD * 8);
    const __bf16* a1s = bases_h + (size_t)r1 * (FD * 8);

    const int col = nBase + li;
    const __bf16* w3b = bw3 + (size_t)col * FD;
    const __bf16* w3s = sw3 + (size_t)col * (FD * 8);

    v8f acca = {}, accb = {};
    for (int kb = 0; kb < FD; kb += 32) {
        const int ko = kb + g8;
        v16bf b = load_frag(w3b, ko);
        acca = wmma_bf16(load_frag(a0b, ko), b, acca);
        accb = wmma_bf16(load_frag(a1b, ko), b, accb);
    }
    for (int kb = 0; kb < FD * 8; kb += 32) {
        const int ko = kb + g8;
        v16bf b = load_frag(w3s, ko);
        acca = wmma_bf16(load_frag(a0s, ko), b, acca);
        accb = wmma_bf16(load_frag(a1s, ko), b, accb);
    }
#pragma unroll
    for (int sub = 0; sub < 2; ++sub) {
        const v8f* p = sub ? &accb : &acca;
#pragma unroll
        for (int v = 0; v < 8; ++v) {
            const int rr = mBase + sub * 16 + g8 + v;
            if (rr >= count) continue;
            const int token = rows_e[rr];
            const float c = coef_e[rr];
            atomicAdd(out + (size_t)token * HD + col, c * (*p)[v]);
        }
    }
}

extern "C" void kernel_launch(void* const* d_in, const int* in_sizes, int n_in,
                              void* d_out, int out_size, void* d_ws, size_t ws_size,
                              hipStream_t stream) {
    (void)in_sizes; (void)n_in; (void)ws_size;
    const float* x   = (const float*)d_in[0];
    const float* gw  = (const float*)d_in[1];
    const float* bw1 = (const float*)d_in[2];
    const float* sw1 = (const float*)d_in[3];
    const float* bw2 = (const float*)d_in[4];
    const float* sw2 = (const float*)d_in[5];
    const float* bw3 = (const float*)d_in[6];
    const float* sw3 = (const float*)d_in[7];
    float* out = (float*)d_out;

    char* ws = (char*)d_ws;
    size_t off = 0;
    auto wsalloc = [&](size_t bytes) -> void* {
        void* p = ws + off;
        off = (off + bytes + 255) & ~(size_t)255;
        return p;
    };
    int*    cnt     = (int*)   wsalloc(NE * sizeof(int));
    int*    rows    = (int*)   wsalloc((size_t)NE * NB * sizeof(int));
    float*  coef    = (float*) wsalloc((size_t)NE * NB * sizeof(float));
    __bf16* silu_x  = (__bf16*)wsalloc((size_t)NB * HD * 2);             //  4 MB
    __bf16* bases_x = (__bf16*)wsalloc((size_t)NB * HD * 8 * 2);         // 32 MB
    __bf16* silu_h  = (__bf16*)wsalloc((size_t)NB * FD * 2);             //  8 MB
    __bf16* bases_h = (__bf16*)wsalloc((size_t)NB * FD * 8 * 2);         // 64 MB
    // per-expert bf16 weight staging (reused across experts)            // 27 MB
    __bf16* wb1 = (__bf16*)wsalloc((size_t)FD * HD * 2);
    __bf16* ws1 = (__bf16*)wsalloc((size_t)FD * HD * 8 * 2);
    __bf16* wb2 = (__bf16*)wsalloc((size_t)FD * HD * 2);
    __bf16* ws2 = (__bf16*)wsalloc((size_t)FD * HD * 8 * 2);
    __bf16* wb3 = (__bf16*)wsalloc((size_t)HD * FD * 2);
    __bf16* ws3 = (__bf16*)wsalloc((size_t)HD * FD * 8 * 2);

    hipMemsetAsync(cnt, 0, NE * sizeof(int), stream);
    hipMemsetAsync(out, 0, (size_t)out_size * sizeof(float), stream);

    gate_kernel<<<NB / 256, 256, 0, stream>>>(x, gw, cnt, rows, coef);
    prep_x_kernel<<<(NB * HD) / 256, 256, 0, stream>>>(x, silu_x, bases_x);

    const int nB8 = FD * HD / 8;         // base-weight elems / 8
    const int nS8 = FD * HD;             // spline-weight elems / 8
    for (int e = 0; e < NE; ++e) {
        cvt_kernel<<<(nB8 + 255) / 256, 256, 0, stream>>>(bw1 + (size_t)e * FD * HD, wb1, nB8);
        cvt_kernel<<<(nS8 + 255) / 256, 256, 0, stream>>>(sw1 + (size_t)e * FD * HD * 8, ws1, nS8);
        cvt_kernel<<<(nB8 + 255) / 256, 256, 0, stream>>>(bw2 + (size_t)e * FD * HD, wb2, nB8);
        cvt_kernel<<<(nS8 + 255) / 256, 256, 0, stream>>>(sw2 + (size_t)e * FD * HD * 8, ws2, nS8);
        kan12_kernel<<<dim3(NB / 64, FD / 64), 256, 0, stream>>>(
            silu_x, bases_x, wb1, ws1, wb2, ws2,
            cnt + e, rows + (size_t)e * NB, silu_h, bases_h);
        cvt_kernel<<<(nB8 + 255) / 256, 256, 0, stream>>>(bw3 + (size_t)e * HD * FD, wb3, nB8);
        cvt_kernel<<<(nS8 + 255) / 256, 256, 0, stream>>>(sw3 + (size_t)e * HD * FD * 8, ws3, nS8);
        kan3_kernel<<<dim3(NB / 64, HD / 64), 256, 0, stream>>>(
            silu_h, bases_h, wb3, ws3,
            cnt + e, rows + (size_t)e * NB, coef + (size_t)e * NB, out);
    }
}
